// AutoEncoderTopK_58815282151849
// MI455X (gfx1250) — matrix operand order repaired and checked by reading, not compiled, not run on hardware
//
#include <hip/hip_runtime.h>
#include <hip/hip_bf16.h>

typedef float  v8f   __attribute__((ext_vector_type(8)));
typedef __bf16 v16bf __attribute__((ext_vector_type(16)));
typedef __bf16 v8bf  __attribute__((ext_vector_type(8)));

#define TM 64    // block tile M (rows of x)
#define TN 128   // block tile N (features)
#define TK 32    // K chunk == WMMA bf16 K
#define LDK 40   // padded bf16 row stride (bank-conflict-free, 16B aligned runs)

// -------- helpers ---------------------------------------------------------
static __device__ inline v16bf frag16(const __bf16* p0, const __bf16* p1) {
  v8bf a = *(const v8bf*)p0;
  v8bf b = *(const v8bf*)p1;
  v16bf r;
#pragma unroll
  for (int e = 0; e < 8; ++e) { r[e] = a[e]; r[8 + e] = b[e]; }
  return r;
}

// -------- 1. transpose W_dec [D,F] -> W_decT [F,D] ------------------------
__global__ __launch_bounds__(256) void transpose_kernel(
    const float* __restrict__ Wdec, float* __restrict__ WdT, int D, int F) {
  __shared__ float tile[32][33];
  const int f0 = blockIdx.x * 32;
  const int d0 = blockIdx.y * 32;
  const int tx = threadIdx.x;   // 0..31
  const int ty = threadIdx.y;   // 0..7
#pragma unroll
  for (int j = 0; j < 4; ++j)
    tile[ty + j * 8][tx] = Wdec[(size_t)(d0 + ty + j * 8) * F + f0 + tx];
  __syncthreads();
#pragma unroll
  for (int j = 0; j < 4; ++j)
    WdT[(size_t)(f0 + ty + j * 8) * D + d0 + tx] = tile[tx][ty + j * 8];
}

// -------- 2. encoder: acts = relu((x - b_dec) @ W_enc^T + b_enc) ----------
// Split-bf16 (hi/lo) GEMM: three v_wmma_f32_16x16x32_bf16 per K-chunk give
// ~fp32 accuracy with fp32 accumulation.
__global__ __launch_bounds__(256) void encode_kernel(
    const float* __restrict__ x, const float* __restrict__ Wenc,
    const float* __restrict__ benc, const float* __restrict__ bdec,
    float* __restrict__ acts, int B, int D, int F) {
  __shared__ __bf16 sAh[TM * LDK];
  __shared__ __bf16 sAl[TM * LDK];
  __shared__ __bf16 sBh[TN * LDK];
  __shared__ __bf16 sBl[TN * LDK];

  const int tid  = threadIdx.x;
  const int lane = tid & 31;
  const int wid  = tid >> 5;   // 0..7
  const int l    = lane & 15;  // N / M index within 16
  const int h    = lane >> 4;  // half-wave selector

  const int bm = blockIdx.y * TM;
  const int bn = blockIdx.x * TN;
  const int m0 = (wid & 1) * 32;   // wave sub-tile origin (M)
  const int n0 = (wid >> 1) * 32;  // wave sub-tile origin (N)

  v8f acc[2][2];
#pragma unroll
  for (int i = 0; i < 2; ++i)
#pragma unroll
    for (int j = 0; j < 2; ++j)
#pragma unroll
      for (int r = 0; r < 8; ++r) acc[i][j][r] = 0.0f;

  for (int k0 = 0; k0 < D; k0 += TK) {
    // ---- stage A: (x - b_dec), 64x32 fp32 -> hi/lo bf16, [m][k] layout
#pragma unroll
    for (int p = 0; p < 2; ++p) {
      const int idx = tid + p * 256;         // 512 float4 total
      const int row = idx >> 3;
      const int c4  = (idx & 7) * 4;
      const float4 xv = *(const float4*)(x + (size_t)(bm + row) * D + k0 + c4);
      const float4 bd = *(const float4*)(bdec + k0 + c4);
      const float* xp = (const float*)&xv;
      const float* bp = (const float*)&bd;
#pragma unroll
      for (int q = 0; q < 4; ++q) {
        const float v = xp[q] - bp[q];
        const __bf16 hi = (__bf16)v;
        sAh[row * LDK + c4 + q] = hi;
        sAl[row * LDK + c4 + q] = (__bf16)(v - (float)hi);
      }
    }
    // ---- stage B: W_enc rows are already [n][k] -> no transpose needed
#pragma unroll
    for (int p = 0; p < 4; ++p) {
      const int idx = tid + p * 256;         // 1024 float4 total
      const int n   = idx >> 3;
      const int c4  = (idx & 7) * 4;
      const float4 wv = *(const float4*)(Wenc + (size_t)(bn + n) * D + k0 + c4);
      const float* wp = (const float*)&wv;
#pragma unroll
      for (int q = 0; q < 4; ++q) {
        const float v = wp[q];
        const __bf16 hi = (__bf16)v;
        sBh[n * LDK + c4 + q] = hi;
        sBl[n * LDK + c4 + q] = (__bf16)(v - (float)hi);
      }
    }
    if (k0 + TK < D)  // gfx1250 global_prefetch_b8 for the next B panel
      __builtin_prefetch(Wenc + (size_t)(bn + (tid >> 1)) * D + k0 + TK, 0, 0);
    __syncthreads();

    // ---- fragments (ISA §7.12.2 layouts)
    v16bf ahi[2], alo[2], bhi[2], blo[2];
#pragma unroll
    for (int i = 0; i < 2; ++i) {
      const int mo = (m0 + i * 16 + l) * LDK;
      ahi[i] = frag16(sAh + mo + h * 8, sAh + mo + 16 + h * 8);
      alo[i] = frag16(sAl + mo + h * 8, sAl + mo + 16 + h * 8);
    }
#pragma unroll
    for (int j = 0; j < 2; ++j) {
      const int no = (n0 + j * 16 + l) * LDK;
      bhi[j] = frag16(sBh + no + h * 8, sBh + no + 16 + h * 8);
      blo[j] = frag16(sBl + no + h * 8, sBl + no + 16 + h * 8);
    }
#pragma unroll
    for (int i = 0; i < 2; ++i)
#pragma unroll
      for (int j = 0; j < 2; ++j) {
        acc[i][j] = __builtin_amdgcn_wmma_f32_16x16x32_bf16(
            false, ahi[i], false, bhi[j], (short)0, acc[i][j], false, false);
        acc[i][j] = __builtin_amdgcn_wmma_f32_16x16x32_bf16(
            false, alo[i], false, bhi[j], (short)0, acc[i][j], false, false);
        acc[i][j] = __builtin_amdgcn_wmma_f32_16x16x32_bf16(
            false, ahi[i], false, blo[j], (short)0, acc[i][j], false, false);
      }
    __syncthreads();
  }

  // ---- epilogue: + b_enc, relu, store (C/D layout: M = 8*h + r, N = l)
#pragma unroll
  for (int i = 0; i < 2; ++i)
#pragma unroll
    for (int j = 0; j < 2; ++j) {
      const int f = bn + n0 + j * 16 + l;
      const float be = benc[f];
#pragma unroll
      for (int r = 0; r < 8; ++r) {
        const int m = bm + m0 + i * 16 + h * 8 + r;
        acts[(size_t)m * F + f] = fmaxf(acc[i][j][r] + be, 0.0f);
      }
    }
}

// -------- 3. exact per-row top-K via 4-pass radix select ------------------
#define TOPK 64
__global__ __launch_bounds__(256) void topk_kernel(
    const float* __restrict__ acts, float* __restrict__ tv,
    unsigned* __restrict__ ti, int F) {
  const int b = blockIdx.x;
  const float* row = acts + (size_t)b * F;
  const int tid = threadIdx.x;

  __shared__ unsigned hist[256];
  __shared__ unsigned sprefix;
  __shared__ int srem;
  __shared__ unsigned scnt_gt;
  __shared__ int sgt_total;
  __shared__ unsigned scnt_eq;

  if (tid == 0) { sprefix = 0; srem = TOPK; }

  for (int pass = 0; pass < 4; ++pass) {
    const int shift = 24 - 8 * pass;
    hist[tid] = 0;
    __syncthreads();
    const unsigned pref = sprefix;
    for (int f = tid; f < F; f += 256) {
      const unsigned u = __float_as_uint(row[f]);  // relu'd => uint order
      if (pass == 0 || (u >> (shift + 8)) == pref)
        atomicAdd(&hist[(u >> shift) & 255u], 1u);
    }
    __syncthreads();
    if (tid == 0) {
      int rem = srem;
      unsigned bin = 0;
      for (int i = 255; i >= 0; --i) {
        const int c = (int)hist[i];
        if (c >= rem) { bin = (unsigned)i; break; }
        rem -= c;
      }
      srem = rem;
      sprefix = (sprefix << 8) | bin;
    }
    __syncthreads();
  }
  const unsigned kth = sprefix;  // bit pattern of the K-th largest value

  if (tid == 0) scnt_gt = 0;
  __syncthreads();
  for (int f = tid; f < F; f += 256) {
    const float v = row[f];
    if (__float_as_uint(v) > kth) {
      const unsigned p = atomicAdd(&scnt_gt, 1u);
      tv[(size_t)b * TOPK + p] = v;
      ti[(size_t)b * TOPK + p] = (unsigned)f;
    }
  }
  __syncthreads();
  if (tid == 0) { sgt_total = (int)scnt_gt; scnt_eq = 0; }
  __syncthreads();
  const int base = sgt_total;
  for (int f = tid; f < F; f += 256) {
    const float v = row[f];
    if (__float_as_uint(v) == kth) {
      const int p = base + (int)atomicAdd(&scnt_eq, 1u);
      if (p < TOPK) {
        tv[(size_t)b * TOPK + p] = v;
        ti[(size_t)b * TOPK + p] = (unsigned)f;
      }
    }
  }
}

// -------- 4. sparse decode: x_hat = b_dec + sum_k v_k * W_decT[idx_k,:] ---
__global__ __launch_bounds__(256) void decode_kernel(
    const float* __restrict__ tv, const unsigned* __restrict__ ti,
    const float* __restrict__ WdT, const float* __restrict__ bdec,
    float* __restrict__ out, int D) {
  const int b = blockIdx.x;
  __shared__ float sv[TOPK];
  __shared__ unsigned si[TOPK];
  const int tid = threadIdx.x;
  if (tid < TOPK) {
    sv[tid] = tv[(size_t)b * TOPK + tid];
    si[tid] = ti[(size_t)b * TOPK + tid];
  }
  __syncthreads();
  for (int d = tid; d < D; d += 256) {
    float acc = bdec[d];
#pragma unroll 4
    for (int k = 0; k < TOPK; ++k) {
      if ((k & 15) == 0 && k + 16 < TOPK)
        __builtin_prefetch(WdT + (size_t)si[k + 16] * D + d, 0, 0);
      acc += sv[k] * WdT[(size_t)si[k] * D + d];
    }
    out[(size_t)b * D + d] = acc;
  }
}

// -------- launch ----------------------------------------------------------
extern "C" void kernel_launch(void* const* d_in, const int* in_sizes, int n_in,
                              void* d_out, int out_size, void* d_ws, size_t ws_size,
                              hipStream_t stream) {
  const float* x    = (const float*)d_in[0];
  const float* Wenc = (const float*)d_in[1];
  const float* benc = (const float*)d_in[2];
  const float* Wdec = (const float*)d_in[3];
  const float* bdec = (const float*)d_in[4];
  float* out = (float*)d_out;

  const int D = in_sizes[4];            // 2304
  const int F = in_sizes[2];            // 36864
  const int B = in_sizes[0] / D;        // 8192

  char* ws = (char*)d_ws;
  float*    acts = (float*)ws;                                   // B*F fp32
  float*    WdT  = (float*)(ws + (size_t)B * F * 4);             // F*D fp32
  float*    tv   = (float*)(ws + (size_t)B * F * 4 + (size_t)F * D * 4);
  unsigned* ti   = (unsigned*)((char*)tv + (size_t)B * TOPK * 4);

  // 1. transpose W_dec for coalesced row gathers in decode
  transpose_kernel<<<dim3(F / 32, D / 32), dim3(32, 8), 0, stream>>>(Wdec, WdT, D, F);
  // 2. encoder GEMM (split-bf16 WMMA) + bias + relu
  encode_kernel<<<dim3(F / TN, B / TM), 256, 0, stream>>>(x, Wenc, benc, bdec, acts, B, D, F);
  // 3. exact top-64 per row
  topk_kernel<<<B, 256, 0, stream>>>(acts, tv, ti, F);
  // 4. sparse decode
  decode_kernel<<<B, 256, 0, stream>>>(tv, ti, WdT, bdec, out, D);
}
